// UnifiedSelfAttention_65481071404706
// MI455X (gfx1250) — matrix operand order repaired and checked by reading, not compiled, not run on hardware
//
#include <hip/hip_runtime.h>

// ---------------------------------------------------------------------------
// Fused self-attention for MI455X (gfx1250, wave32, WMMA bf16 -> f32 acc)
// B=2, S=2048, E=1024, H=16, Dh=64
// Round 2: cooperative async-to-LDS double-buffered K/V staging in attention.
// ---------------------------------------------------------------------------

typedef __attribute__((ext_vector_type(16))) __bf16 v16bf;
typedef __attribute__((ext_vector_type(8)))  __bf16 v8bf;
typedef __attribute__((ext_vector_type(8)))  float  v8f;

#define BATCH 2
#define SEQ   2048
#define EMB   1024
#define NH    16
#define DH    64

// Build a 16-element bf16 fragment from two 8-element (16-byte) loads.
// Works for both global and LDS pointers (addrspace inferred by the compiler).
__device__ __forceinline__ v16bf load_frag16(const __bf16* p, int off1) {
    v8bf x0 = *(const v8bf*)(p);
    v8bf x1 = *(const v8bf*)(p + off1);
    v16bf r;
#pragma unroll
    for (int j = 0; j < 8; ++j) { r[j] = x0[j]; r[j + 8] = x1[j]; }
    return r;
}

// One per-lane async 16-byte copy global -> LDS (tracked by ASYNCcnt).
__device__ __forceinline__ void async_ld_b128(unsigned lds_off, const void* gptr) {
    asm volatile("global_load_async_to_lds_b128 %0, %1, off"
                 :: "v"(lds_off), "v"(gptr)
                 : "memory");
}

// ---------------------------------------------------------------------------
// Kernel 1: fp32 -> bf16 conversion (X and the three weight matrices)
// ---------------------------------------------------------------------------
__global__ void cvt_f32_to_bf16(const float* __restrict__ in,
                                __bf16* __restrict__ out, int n) {
    int i = (blockIdx.x * blockDim.x + threadIdx.x) * 8;
    if (i + 8 <= n) {
        v8bf o;
#pragma unroll
        for (int j = 0; j < 8; ++j) o[j] = (__bf16)in[i + j];
        *(v8bf*)(out + i) = o;
    }
}

// ---------------------------------------------------------------------------
// Kernel 2: projection GEMM  out(m,n) = sum_k X[m,k] * W[n,k]   (x @ W.T)
// One wave -> 16x64 output tile (A fragment reused across 4 WMMAs per k-step).
// mode==0: store [B,H,S,Dh] (Q,K).  mode==1: store transposed [B,H,Dh,S] (V).
// ---------------------------------------------------------------------------
__global__ __launch_bounds__(256)
void qkv_proj_wmma(const __bf16* __restrict__ X, const __bf16* __restrict__ W,
                   __bf16* __restrict__ out, int transposed) {
    const int lane = threadIdx.x & 31;
    const int wid  = threadIdx.x >> 5;
    const int hi   = lane >> 4;      // lane half (0/1)
    const int lo   = lane & 15;

    const int m0 = blockIdx.x * 16;                  // row tile: [0,4096)
    const int n0 = (blockIdx.y * 8 + wid) * 64;      // col tile: [0,1024)

    v8f acc[4] = {};                                  // 16x64 f32 accumulators

    const __bf16* arow_base = X + (size_t)(m0 + lo) * EMB;

    for (int k = 0; k < EMB; k += 32) {
        // A fragment (16x32): lane row = lo; K runs at {hi*8, hi*8+16}
        v16bf a = load_frag16(arow_base + k + hi * 8, 16);
#pragma unroll
        for (int nt = 0; nt < 4; ++nt) {
            // B fragment (32x16): column n = n0+nt*16+lo is row n of W;
            // K elements at {k + hi*8, k + hi*8 + 16} (two runs of 8 bf16)
            const __bf16* brow = W + (size_t)(n0 + nt * 16 + lo) * EMB + k + hi * 8;
            v16bf b = load_frag16(brow, 16);
            acc[nt] = __builtin_amdgcn_wmma_f32_16x16x32_bf16(
                false, a, false, b, (short)0, acc[nt], false, false);
        }
    }

    // ---- store ----
    const int bidx = m0 / SEQ;            // batch
    const int srow = (m0 % SEQ) + 8 * hi; // first seq row this lane-half owns
    const int head = n0 / DH;             // n0 is a multiple of 64

    if (!transposed) {
        // Q/K: [B,H,S,Dh]
#pragma unroll
        for (int nt = 0; nt < 4; ++nt) {
            const int d = nt * 16 + lo;
            __bf16* dst = out + (((size_t)(bidx * NH + head) * SEQ) + srow) * DH + d;
#pragma unroll
            for (int v = 0; v < 8; ++v)
                dst[(size_t)v * DH] = (__bf16)acc[nt][v];
        }
    } else {
        // V^T: [B,H,Dh,S] -> 8 contiguous seq positions per lane = one b128 store
#pragma unroll
        for (int nt = 0; nt < 4; ++nt) {
            const int d = nt * 16 + lo;
            v8bf o;
#pragma unroll
            for (int v = 0; v < 8; ++v) o[v] = (__bf16)acc[nt][v];
            *(v8bf*)(out + ((size_t)(bidx * NH + head) * DH + d) * SEQ + srow) = o;
        }
    }
}

// ---------------------------------------------------------------------------
// Kernel 3: flash attention.  blockIdx.y = (b*NH+h); all 8 waves of a block
// share the same K/V stream, so K (32x64) and V^T (64x32) tiles are staged
// into LDS with async-to-LDS loads (double buffered, one b128 per thread per
// tile), synchronized via s_wait_asynccnt + block barrier.  Per 32-key step:
// 4 score WMMAs, f32 online softmax, P staged via per-wave LDS, 4 PV WMMAs.
// ---------------------------------------------------------------------------
__global__ __launch_bounds__(256)
void flash_attn_wmma(const __bf16* __restrict__ Q, const __bf16* __restrict__ K,
                     const __bf16* __restrict__ Vt, float* __restrict__ out) {
    __shared__ __bf16 kbuf[2][32 * DH];   // 2 x 4KB   key rows, Dh contiguous
    __shared__ __bf16 vbuf[2][DH * 32];   // 2 x 4KB   d rows, keys contiguous
    __shared__ __bf16 plds[8][16 * 48];   // 12KB      per-wave P staging

    const int tid  = threadIdx.x;
    const int lane = tid & 31;
    const int wid  = tid >> 5;
    const int hi   = lane >> 4;
    const int lo   = lane & 15;

    const int bh   = blockIdx.y;                       // b*NH + h
    const int bidx = bh / NH;
    const int head = bh % NH;
    const int q0   = (blockIdx.x * 8 + wid) * 16;      // query tile start

    const __bf16* Qb = Q  + (size_t)bh * SEQ * DH;
    const __bf16* Kb = K  + (size_t)bh * SEQ * DH;
    const __bf16* Vb = Vt + (size_t)bh * DH * SEQ;
    __bf16* pl = plds[wid];

    // Cooperative async stage of one 32-key K tile + V^T tile into buffer b.
    // 256 threads x 16B = 4KB per tile: exactly one async b128 each.
    auto stage = [&](int b, int kb) {
        {   // K tile: thread t -> key row t>>3, 16B chunk t&7 (row = 128B)
            const int row = tid >> 3, ch = tid & 7;
            unsigned loff = (unsigned)(size_t)&kbuf[b][row * DH] + ch * 16;
            const void* g = (const char*)(Kb + (size_t)(kb + row) * DH) + ch * 16;
            async_ld_b128(loff, g);
        }
        {   // V^T tile: thread t -> d row t>>2, 16B chunk t&3 (row = 64B)
            const int dr = tid >> 2, ch = tid & 3;
            unsigned loff = (unsigned)(size_t)&vbuf[b][dr * 32] + ch * 16;
            const void* g = (const char*)(Vb + (size_t)dr * SEQ + kb) + ch * 16;
            async_ld_b128(loff, g);
        }
    };

    // Q A-fragments for the two Dh chunks (live in registers for all keys)
    v16bf qf[2];
#pragma unroll
    for (int c = 0; c < 2; ++c)
        qf[c] = load_frag16(Qb + (size_t)(q0 + lo) * DH + c * 32 + hi * 8, 16);

    float mrow[8], lrow[8];
    v8f o[4] = {};
#pragma unroll
    for (int v = 0; v < 8; ++v) { mrow[v] = -3.0e38f; lrow[v] = 0.0f; }

    const float scale = 0.125f;   // 1/sqrt(Dh)

    // prologue: fill buffer 0
    stage(0, 0);
    asm volatile("s_wait_asynccnt 0x0" ::: "memory");
    __syncthreads();

    int cur = 0;
    for (int kb = 0; kb < SEQ; kb += 32) {
        // kick off the next tile while we compute on the current one
        if (kb + 32 < SEQ) stage(cur ^ 1, kb + 32);

        const __bf16* kt = kbuf[cur];
        const __bf16* vt = vbuf[cur];

        // ---- scores: two 16x16 tiles covering keys [kb, kb+32) ----
        v8f st[2] = {};
#pragma unroll
        for (int t = 0; t < 2; ++t) {
#pragma unroll
            for (int c = 0; c < 2; ++c) {
                // B frag: key = local col, Dh elements contiguous in LDS row
                const __bf16* kr = kt + (t * 16 + lo) * DH + c * 32 + hi * 8;
                v16bf bfr = load_frag16(kr, 16);
                st[t] = __builtin_amdgcn_wmma_f32_16x16x32_bf16(
                    false, qf[c], false, bfr, (short)0, st[t], false, false);
            }
        }

        // ---- online softmax (rows live at vgpr v, lane-half hi) ----
        float rmax[8];
#pragma unroll
        for (int v = 0; v < 8; ++v) {
            st[0][v] *= scale; st[1][v] *= scale;
            rmax[v] = fmaxf(st[0][v], st[1][v]);
        }
#pragma unroll
        for (int m = 1; m < 16; m <<= 1)
#pragma unroll
            for (int v = 0; v < 8; ++v)
                rmax[v] = fmaxf(rmax[v], __shfl_xor(rmax[v], m, 32));

        float alpha[8], rsum[8];
#pragma unroll
        for (int v = 0; v < 8; ++v) {
            float mn = fmaxf(mrow[v], rmax[v]);
            alpha[v] = __expf(mrow[v] - mn);
            mrow[v]  = mn;
            st[0][v] = __expf(st[0][v] - mn);
            st[1][v] = __expf(st[1][v] - mn);
            rsum[v]  = st[0][v] + st[1][v];
        }
#pragma unroll
        for (int m = 1; m < 16; m <<= 1)
#pragma unroll
            for (int v = 0; v < 8; ++v)
                rsum[v] += __shfl_xor(rsum[v], m, 32);
#pragma unroll
        for (int v = 0; v < 8; ++v)
            lrow[v] = lrow[v] * alpha[v] + rsum[v];
#pragma unroll
        for (int nt = 0; nt < 4; ++nt)
#pragma unroll
            for (int v = 0; v < 8; ++v)
                o[nt][v] *= alpha[v];

        // ---- stage P into per-wave LDS (C/D layout -> row-major bf16) ----
#pragma unroll
        for (int v = 0; v < 8; ++v) {
            int r = v + 8 * hi;
            pl[r * 48 + lo]      = (__bf16)st[0][v];
            pl[r * 48 + 16 + lo] = (__bf16)st[1][v];
        }
        asm volatile("s_wait_dscnt 0x0" ::: "memory");

        // ---- read P back as a 16x32 A-fragment ----
        v16bf pf = load_frag16(pl + lo * 48 + hi * 8, 16);

        // ---- PV: 4 WMMAs across Dh (V^T rows contiguous along keys) ----
#pragma unroll
        for (int nt = 0; nt < 4; ++nt) {
            const __bf16* vr = vt + (nt * 16 + lo) * 32 + hi * 8;
            v16bf bfr = load_frag16(vr, 16);
            o[nt] = __builtin_amdgcn_wmma_f32_16x16x32_bf16(
                false, pf, false, bfr, (short)0, o[nt], false, false);
        }

        // next buffer must be fully written; current one fully read by all
        asm volatile("s_wait_asynccnt 0x0" ::: "memory");
        __syncthreads();
        cur ^= 1;
    }

    // ---- epilogue: normalize and store fp32 [B,S,E] with d = h*64 + d ----
#pragma unroll
    for (int nt = 0; nt < 4; ++nt) {
        const int d = head * DH + nt * 16 + lo;
#pragma unroll
        for (int v = 0; v < 8; ++v) {
            const int s = q0 + v + 8 * hi;
            out[((size_t)bidx * SEQ + s) * EMB + d] = o[nt][v] * (1.0f / lrow[v]);
        }
    }
}

// ---------------------------------------------------------------------------
// Launcher
// ---------------------------------------------------------------------------
extern "C" void kernel_launch(void* const* d_in, const int* in_sizes, int n_in,
                              void* d_out, int out_size, void* d_ws, size_t ws_size,
                              hipStream_t stream) {
    const float* X  = (const float*)d_in[0];
    const float* Wq = (const float*)d_in[1];
    const float* Wk = (const float*)d_in[2];
    const float* Wv = (const float*)d_in[3];
    float* out = (float*)d_out;

    const size_t nX = (size_t)BATCH * SEQ * EMB;   // 4,194,304
    const size_t nW = (size_t)EMB * EMB;           // 1,048,576

    char* ws = (char*)d_ws;
    __bf16* Xbf  = (__bf16*)ws;  ws += nX * 2;
    __bf16* Wqbf = (__bf16*)ws;  ws += nW * 2;
    __bf16* Wkbf = (__bf16*)ws;  ws += nW * 2;
    __bf16* Wvbf = (__bf16*)ws;  ws += nW * 2;
    __bf16* Qbf  = (__bf16*)ws;  ws += nX * 2;
    __bf16* Kbf  = (__bf16*)ws;  ws += nX * 2;
    __bf16* Vtbf = (__bf16*)ws;  ws += nX * 2;

    // 1) fp32 -> bf16
    cvt_f32_to_bf16<<<(int)(nX / 8 / 256), 256, 0, stream>>>(X,  Xbf,  (int)nX);
    cvt_f32_to_bf16<<<(int)(nW / 8 / 256), 256, 0, stream>>>(Wq, Wqbf, (int)nW);
    cvt_f32_to_bf16<<<(int)(nW / 8 / 256), 256, 0, stream>>>(Wk, Wkbf, (int)nW);
    cvt_f32_to_bf16<<<(int)(nW / 8 / 256), 256, 0, stream>>>(Wv, Wvbf, (int)nW);

    // 2) projections: grid = (4096/16 row tiles, 1024/(64*8) col groups)
    dim3 pg(BATCH * SEQ / 16, EMB / (64 * 8));
    qkv_proj_wmma<<<pg, 256, 0, stream>>>(Xbf, Wqbf, Qbf,  0);
    qkv_proj_wmma<<<pg, 256, 0, stream>>>(Xbf, Wkbf, Kbf,  0);
    qkv_proj_wmma<<<pg, 256, 0, stream>>>(Xbf, Wvbf, Vtbf, 1);

    // 3) flash attention: grid = (2048/(16*8) q-groups, B*H)
    flash_attn_wmma<<<dim3(SEQ / 128, BATCH * NH), 256, 0, stream>>>(
        Qbf, Kbf, Vtbf, out);
}
